// FastGraphWaveLayer_57604101374355
// MI455X (gfx1250) — matrix operand order repaired and checked by reading, not compiled, not run on hardware
//
#include <hip/hip_runtime.h>
#include <math.h>

// ---------------- CDNA5 WMMA types ----------------
typedef __attribute__((ext_vector_type(16))) __bf16 v16bf;
typedef __attribute__((ext_vector_type(8)))  float  v8f;

union Frag {
    v16bf v;
    uint4 q[2];
};

union Pack8 {
    uint4 q;
    unsigned short s[8];
};

__device__ __forceinline__ unsigned short f2bf(float f) {
    unsigned int u = __float_as_uint(f);
    u += 0x7FFFu + ((u >> 16) & 1u);   // round-to-nearest-even
    return (unsigned short)(u >> 16);
}

#define N_B    8
#define N_T    256
#define N_NN   512
#define N_F    6
#define N_KS   5
#define N_E    65536
#define N_NODE 4096            // B*NN
#define N_EE   69632           // E + self loops
#define PLANE  (2048 * 512)    // B*T*NN elements per feature plane
#define WMAT   (512 * 512)     // one conv weight matrix
#define QMAT   (256 * 256)     // one qkv weight matrix

// =====================================================================
// Kernel 0a: x [B,T,NN,F] f32  ->  per-feature planar bf16 [F][B*T][NN]
// =====================================================================
__global__ __launch_bounds__(256) void xplane_kernel(
    const float* __restrict__ x, unsigned short* __restrict__ xp)
{
    const int idx = blockIdx.x * 256 + threadIdx.x;   // over B*T*NN
    const float* s = x + (size_t)idx * N_F;
    #pragma unroll
    for (int f = 0; f < N_F; ++f)
        xp[(size_t)f * PLANE + idx] = f2bf(s[f]);
}

// =====================================================================
// Kernel 0b: conv weights f32 [cv][f][k][c][o] -> bf16 transposed [zz][o][c]
//   zz = (cv*6+f)*5+k ; LDS-tiled 32x32 transpose, coalesced both sides
// =====================================================================
__global__ __launch_bounds__(256) void wprep_kernel(
    const float* __restrict__ caw, const float* __restrict__ cbw,
    unsigned short* __restrict__ wt)
{
    __shared__ unsigned short tile[32][33];
    const int zz  = blockIdx.z;              // 0..59
    const int cv  = zz / 30;
    const int rem = zz % 30;                 // f*5+k
    const float* src = (cv ? cbw : caw) + (size_t)rem * WMAT;
    unsigned short* dst = wt + (size_t)zz * WMAT;
    const int o0 = blockIdx.x * 32;
    const int c0 = blockIdx.y * 32;
    const int tx = threadIdx.x & 31;
    const int ty = threadIdx.x >> 5;         // 0..7
    #pragma unroll
    for (int r = 0; r < 4; ++r) {
        const int row = ty + r * 8;
        tile[row][tx] = f2bf(src[(size_t)(c0 + row) * N_NN + o0 + tx]);
    }
    __syncthreads();
    #pragma unroll
    for (int r = 0; r < 4; ++r) {
        const int row = ty + r * 8;
        dst[(size_t)(o0 + row) * N_NN + c0 + tx] = tile[tx][row];
    }
}

// =====================================================================
// Kernel 0c: Q/K/V weights f32 [m][f][c][o] -> bf16 transposed [zz][o][c]
//   zz = m*6+f
// =====================================================================
__global__ __launch_bounds__(256) void qprep_kernel(
    const float* __restrict__ Wq, const float* __restrict__ Wk,
    const float* __restrict__ Wv, unsigned short* __restrict__ wt)
{
    __shared__ unsigned short tile[32][33];
    const int zz = blockIdx.z;               // 0..17
    const int m  = zz / 6;
    const int f  = zz % 6;
    const float* src = ((m == 0) ? Wq : (m == 1) ? Wk : Wv) + (size_t)f * QMAT;
    unsigned short* dst = wt + (size_t)zz * QMAT;
    const int o0 = blockIdx.x * 32;
    const int c0 = blockIdx.y * 32;
    const int tx = threadIdx.x & 31;
    const int ty = threadIdx.x >> 5;
    #pragma unroll
    for (int r = 0; r < 4; ++r) {
        const int row = ty + r * 8;
        tile[row][tx] = f2bf(src[(size_t)(c0 + row) * N_T + o0 + tx]);
    }
    __syncthreads();
    #pragma unroll
    for (int r = 0; r < 4; ++r) {
        const int row = ty + r * 8;
        dst[(size_t)(o0 + row) * N_T + c0 + tx] = tile[tx][row];
    }
}

// =====================================================================
// Kernel 1: fused causal-conv pair + sigmoid*tanh gate + transpose store
// double-buffered LDS; all staging is pure b128 copies (bf16 everywhere)
// =====================================================================
__global__ __launch_bounds__(256) void conv_gate_kernel(
    const unsigned short* __restrict__ xp,    // planar bf16 [2048,512] (this feature)
    const unsigned short* __restrict__ wta,   // bf16 [KS][o=512][c=512]
    const unsigned short* __restrict__ wtb,
    const float* __restrict__ ba,
    const float* __restrict__ bb,
    unsigned short* __restrict__ xn)          // [4096,256] bf16
{
    __shared__ __align__(16) unsigned short As [2][64 * 32];
    __shared__ __align__(16) unsigned short Wta[2][128 * 32];
    __shared__ __align__(16) unsigned short Wtb[2][128 * 32];

    const int tid  = threadIdx.x;
    const int lane = tid & 31;
    const int wid  = tid >> 5;
    const int wm   = wid & 1;        // 2 waves along M
    const int wn   = wid >> 1;       // 4 waves along N
    const int m0   = blockIdx.x * 64;
    const int o0   = blockIdx.y * 128;

    const int ar  = tid >> 2;            // A row 0..63
    const int acb = (tid & 3) * 8;       // A col base
    const int wo  = tid >> 1;            // W out-row 0..127
    const int wcb = (tid & 1) * 16;      // W col base
    const int amg = m0 + ar;
    const int ab  = amg >> 8;            // batch
    const int at  = amg & 255;           // time

    v8f zf = {0.f,0.f,0.f,0.f,0.f,0.f,0.f,0.f};
    v8f acc[2][2][2];                     // [conv][msub][nsub]
    for (int cv = 0; cv < 2; ++cv)
        for (int ms = 0; ms < 2; ++ms)
            for (int ns = 0; ns < 2; ++ns) acc[cv][ms][ns] = zf;

    const int NIT = N_KS * 16;            // 5 taps x (512/32) c-chunks

    // ---- prologue: stage chunk 0 (k=0, c0=0) into buffer 0 ----
    {
        const int ts = at - (N_KS - 1);
        uint4 az = {0u,0u,0u,0u};
        if (ts >= 0)
            az = *reinterpret_cast<const uint4*>(xp + (size_t)(ab * N_T + ts) * N_NN + acb);
        *reinterpret_cast<uint4*>(&As[0][ar * 32 + acb]) = az;
        const unsigned short* wap = wta + (size_t)(o0 + wo) * N_NN + wcb;
        const unsigned short* wbp = wtb + (size_t)(o0 + wo) * N_NN + wcb;
        *reinterpret_cast<uint4*>(&Wta[0][wo * 32 + wcb])     = *reinterpret_cast<const uint4*>(wap);
        *reinterpret_cast<uint4*>(&Wta[0][wo * 32 + wcb + 8]) = *reinterpret_cast<const uint4*>(wap + 8);
        *reinterpret_cast<uint4*>(&Wtb[0][wo * 32 + wcb])     = *reinterpret_cast<const uint4*>(wbp);
        *reinterpret_cast<uint4*>(&Wtb[0][wo * 32 + wcb + 8]) = *reinterpret_cast<const uint4*>(wbp + 8);
    }
    __syncthreads();

    for (int it = 0; it < NIT; ++it) {
        const int  cur  = it & 1;
        const int  nxt  = cur ^ 1;
        const bool more = (it + 1) < NIT;        // uniform

        // ---- issue next chunk's global loads into registers ----
        uint4 aReg = {0u,0u,0u,0u};
        uint4 wA0, wA1, wB0, wB1;
        if (more) {
            const int k1 = (it + 1) >> 4;
            const int c1 = ((it + 1) & 15) * 32;
            const int ts = at + k1 - (N_KS - 1);
            if (ts >= 0)
                aReg = *reinterpret_cast<const uint4*>(
                    xp + (size_t)(ab * N_T + ts) * N_NN + c1 + acb);
            const unsigned short* wap =
                wta + (size_t)k1 * WMAT + (size_t)(o0 + wo) * N_NN + c1 + wcb;
            const unsigned short* wbp =
                wtb + (size_t)k1 * WMAT + (size_t)(o0 + wo) * N_NN + c1 + wcb;
            wA0 = *reinterpret_cast<const uint4*>(wap);
            wA1 = *reinterpret_cast<const uint4*>(wap + 8);
            wB0 = *reinterpret_cast<const uint4*>(wbp);
            wB1 = *reinterpret_cast<const uint4*>(wbp + 8);
            if (it + 2 < NIT) {                   // prefetch 2 chunks ahead into GL2
                const int k2 = (it + 2) >> 4;
                const int c2 = ((it + 2) & 15) * 32;
                __builtin_prefetch(wta + (size_t)k2 * WMAT + (size_t)(o0 + wo) * N_NN + c2 + wcb, 0, 0);
                __builtin_prefetch(wtb + (size_t)k2 * WMAT + (size_t)(o0 + wo) * N_NN + c2 + wcb, 0, 0);
            }
        }

        // ---- compute on current buffer ----
        {
            Frag af[2], fa[2], fb[2];
            const int mrow = lane & 15;
            const int k8   = (lane >> 4) * 16;   // A K-offset per lane half
            #pragma unroll
            for (int ms = 0; ms < 2; ++ms) {
                const unsigned short* p = &As[cur][(wm * 32 + ms * 16 + mrow) * 32];
                af[ms].q[0] = *reinterpret_cast<const uint4*>(p + k8);
                af[ms].q[1] = *reinterpret_cast<const uint4*>(p + k8 + 8);
            }
            const int ncol = lane & 15;
            const int k16  = (lane >> 4) * 16;   // B K-offset per lane half
            #pragma unroll
            for (int ns = 0; ns < 2; ++ns) {
                const unsigned short* pa = &Wta[cur][(wn * 32 + ns * 16 + ncol) * 32 + k16];
                const unsigned short* pb = &Wtb[cur][(wn * 32 + ns * 16 + ncol) * 32 + k16];
                fa[ns].q[0] = *reinterpret_cast<const uint4*>(pa);
                fa[ns].q[1] = *reinterpret_cast<const uint4*>(pa + 8);
                fb[ns].q[0] = *reinterpret_cast<const uint4*>(pb);
                fb[ns].q[1] = *reinterpret_cast<const uint4*>(pb + 8);
            }
            #pragma unroll
            for (int ms = 0; ms < 2; ++ms)
                #pragma unroll
                for (int ns = 0; ns < 2; ++ns) {
                    acc[0][ms][ns] = __builtin_amdgcn_wmma_f32_16x16x32_bf16(
                        false, af[ms].v, false, fa[ns].v, (short)0, acc[0][ms][ns], false, false);
                    acc[1][ms][ns] = __builtin_amdgcn_wmma_f32_16x16x32_bf16(
                        false, af[ms].v, false, fb[ns].v, (short)0, acc[1][ms][ns], false, false);
                }
        }

        // ---- write staged registers into the other buffer ----
        if (more) {
            *reinterpret_cast<uint4*>(&As[nxt][ar * 32 + acb]) = aReg;
            *reinterpret_cast<uint4*>(&Wta[nxt][wo * 32 + wcb])     = wA0;
            *reinterpret_cast<uint4*>(&Wta[nxt][wo * 32 + wcb + 8]) = wA1;
            *reinterpret_cast<uint4*>(&Wtb[nxt][wo * 32 + wcb])     = wB0;
            *reinterpret_cast<uint4*>(&Wtb[nxt][wo * 32 + wcb + 8]) = wB1;
        }
        __syncthreads();
    }

    // ---- epilogue: bias, gate, packed transposed bf16 store ----
    const int ncol = lane & 15;
    const int mhi  = (lane >> 4) * 8;
    for (int ms = 0; ms < 2; ++ms)
        for (int ns = 0; ns < 2; ++ns) {
            const int o = o0 + wn * 32 + ns * 16 + ncol;
            const float biasA = ba[o];
            const float biasB = bb[o];
            const int mg0 = m0 + wm * 32 + ms * 16 + mhi;
            const int b   = mg0 >> 8;
            const int tb  = mg0 & 255;           // 8 consecutive t values
            Pack8 pk;
            #pragma unroll
            for (int r = 0; r < 8; ++r) {
                const float pa = acc[0][ms][ns][r] + biasA;
                const float pb = acc[1][ms][ns][r] + biasB;
                const float g  = (1.f / (1.f + expf(-pa))) * tanhf(pb);
                pk.s[r] = f2bf(g);
            }
            *reinterpret_cast<uint4*>(xn + ((size_t)b * N_NN + o) * N_T + tb) = pk.q;
        }
}

// =====================================================================
// Kernel 2: Q/K/V GEMMs  (grid.z: 0=Q,1=K,2=V), double-buffered
// =====================================================================
__global__ __launch_bounds__(256) void qkv_kernel(
    const unsigned short* __restrict__ xn,
    const unsigned short* __restrict__ wtQ,   // bf16 [o=256][c=256]
    const unsigned short* __restrict__ wtK,
    const unsigned short* __restrict__ wtV,
    const float* __restrict__ bq, const float* __restrict__ bk,
    float* __restrict__ Qo, float* __restrict__ Ko, float* __restrict__ Vo)
{
    __shared__ __align__(16) unsigned short As[2][64 * 32];
    __shared__ __align__(16) unsigned short Wt[2][128 * 32];

    const int z = blockIdx.z;
    const unsigned short* W = (z == 0) ? wtQ : (z == 1) ? wtK : wtV;
    const float* bi = (z == 0) ? bq : (z == 1) ? bk : nullptr;
    float*       Oo = (z == 0) ? Qo : (z == 1) ? Ko : Vo;

    const int tid  = threadIdx.x;
    const int lane = tid & 31;
    const int wid  = tid >> 5;
    const int wm   = wid & 1;
    const int wn   = wid >> 1;
    const int m0   = blockIdx.x * 64;   // node tile
    const int o0   = blockIdx.y * 128;  // out-col tile

    const int ar  = tid >> 2;
    const int acb = (tid & 3) * 8;
    const int wo  = tid >> 1;
    const int wcb = (tid & 1) * 16;

    v8f zf = {0.f,0.f,0.f,0.f,0.f,0.f,0.f,0.f};
    v8f acc[2][2];
    for (int ms = 0; ms < 2; ++ms)
        for (int ns = 0; ns < 2; ++ns) acc[ms][ns] = zf;

    // prologue: stage chunk 0
    {
        *reinterpret_cast<uint4*>(&As[0][ar * 32 + acb]) =
            *reinterpret_cast<const uint4*>(xn + (size_t)(m0 + ar) * N_T + acb);
        const unsigned short* wp = W + (size_t)(o0 + wo) * N_T + wcb;
        *reinterpret_cast<uint4*>(&Wt[0][wo * 32 + wcb])     = *reinterpret_cast<const uint4*>(wp);
        *reinterpret_cast<uint4*>(&Wt[0][wo * 32 + wcb + 8]) = *reinterpret_cast<const uint4*>(wp + 8);
    }
    __syncthreads();

    const int NIT = N_T / 32;   // 8
    for (int it = 0; it < NIT; ++it) {
        const int  cur  = it & 1;
        const int  nxt  = cur ^ 1;
        const bool more = (it + 1) < NIT;

        uint4 aReg = {0u,0u,0u,0u};
        uint4 w0, w1;
        if (more) {
            const int c1 = (it + 1) * 32;
            aReg = *reinterpret_cast<const uint4*>(xn + (size_t)(m0 + ar) * N_T + c1 + acb);
            const unsigned short* wp = W + (size_t)(o0 + wo) * N_T + c1 + wcb;
            w0 = *reinterpret_cast<const uint4*>(wp);
            w1 = *reinterpret_cast<const uint4*>(wp + 8);
        }

        {
            Frag af[2], bf[2];
            const int mrow = lane & 15;
            const int k8   = (lane >> 4) * 16;
            #pragma unroll
            for (int ms = 0; ms < 2; ++ms) {
                const unsigned short* p = &As[cur][(wm * 32 + ms * 16 + mrow) * 32];
                af[ms].q[0] = *reinterpret_cast<const uint4*>(p + k8);
                af[ms].q[1] = *reinterpret_cast<const uint4*>(p + k8 + 8);
            }
            const int ncol = lane & 15;
            const int k16  = (lane >> 4) * 16;
            #pragma unroll
            for (int ns = 0; ns < 2; ++ns) {
                const unsigned short* p = &Wt[cur][(wn * 32 + ns * 16 + ncol) * 32 + k16];
                bf[ns].q[0] = *reinterpret_cast<const uint4*>(p);
                bf[ns].q[1] = *reinterpret_cast<const uint4*>(p + 8);
            }
            #pragma unroll
            for (int ms = 0; ms < 2; ++ms)
                #pragma unroll
                for (int ns = 0; ns < 2; ++ns)
                    acc[ms][ns] = __builtin_amdgcn_wmma_f32_16x16x32_bf16(
                        false, af[ms].v, false, bf[ns].v, (short)0, acc[ms][ns], false, false);
        }

        if (more) {
            *reinterpret_cast<uint4*>(&As[nxt][ar * 32 + acb]) = aReg;
            *reinterpret_cast<uint4*>(&Wt[nxt][wo * 32 + wcb])     = w0;
            *reinterpret_cast<uint4*>(&Wt[nxt][wo * 32 + wcb + 8]) = w1;
        }
        __syncthreads();
    }

    const int ncol = lane & 15;
    const int mhi  = (lane >> 4) * 8;
    for (int ms = 0; ms < 2; ++ms)
        for (int ns = 0; ns < 2; ++ns) {
            const int o = o0 + wn * 32 + ns * 16 + ncol;
            const float bv = bi ? bi[o] : 0.f;
            #pragma unroll
            for (int r = 0; r < 8; ++r) {
                const int node = m0 + wm * 32 + ms * 16 + r + mhi;
                float v = acc[ms][ns][r] + bv;
                if (z < 2) v = (v > 0.f) ? v : 0.2f * v;   // leaky_relu
                Oo[(size_t)node * N_T + o] = v;
            }
        }
}

// =====================================================================
// GAT pipeline
// =====================================================================
__device__ __forceinline__ unsigned int fkey(float f) {
    unsigned int u = __float_as_uint(f);
    return (u & 0x80000000u) ? ~u : (u | 0x80000000u);
}
__device__ __forceinline__ float funkey(unsigned int k) {
    return (k & 0x80000000u) ? __uint_as_float(k & 0x7FFFFFFFu) : __uint_as_float(~k);
}

__global__ __launch_bounds__(256) void gat_init_kernel(
    unsigned int* __restrict__ mkey, float* __restrict__ denom, float* __restrict__ h)
{
    const int idx = blockIdx.x * 256 + threadIdx.x;   // 4096*256 threads
    h[idx] = 0.f;
    if (idx < N_NODE) { mkey[idx] = 0u; denom[idx] = 0.f; }
}

__global__ __launch_bounds__(256) void att_kernel(
    const float* __restrict__ Q, const float* __restrict__ K,
    const int* __restrict__ edges, float* __restrict__ att, unsigned int* __restrict__ mkey)
{
    const int e    = blockIdx.x * 8 + (threadIdx.x >> 5);
    const int lane = threadIdx.x & 31;
    if (e >= N_EE) return;
    int r, c;
    if (e < N_E) { r = edges[e]; c = edges[N_E + e]; } else { r = e - N_E; c = r; }
    const float4* q4 = reinterpret_cast<const float4*>(Q + (size_t)r * N_T);
    const float4* k4 = reinterpret_cast<const float4*>(K + (size_t)c * N_T);
    float s = 0.f;
    #pragma unroll
    for (int j = 0; j < 2; ++j) {
        const float4 a = q4[lane + 32 * j];
        const float4 b = k4[lane + 32 * j];
        s += a.x * b.x + a.y * b.y + a.z * b.z + a.w * b.w;
    }
    for (int off = 16; off > 0; off >>= 1) s += __shfl_down(s, off, 32);
    if (lane == 0) {
        att[e] = s;
        atomicMax(&mkey[r], fkey(s));
    }
}

__global__ __launch_bounds__(256) void expsum_kernel(
    float* __restrict__ att, const unsigned int* __restrict__ mkey,
    float* __restrict__ denom, const int* __restrict__ edges)
{
    const int e = blockIdx.x * 256 + threadIdx.x;
    if (e >= N_EE) return;
    const int r = (e < N_E) ? edges[e] : (e - N_E);
    const float w = expf(att[e] - funkey(mkey[r]));
    att[e] = w;
    atomicAdd(&denom[r], w);
}

__global__ __launch_bounds__(256) void scatter_kernel(
    const float* __restrict__ att, const float* __restrict__ denom,
    const float* __restrict__ V, const int* __restrict__ edges, float* __restrict__ h)
{
    const int e    = blockIdx.x * 8 + (threadIdx.x >> 5);
    const int lane = threadIdx.x & 31;
    if (e >= N_EE) return;
    int r, c;
    if (e < N_E) { r = edges[e]; c = edges[N_E + e]; } else { r = e - N_E; c = r; }
    const float a   = att[e] / denom[r];
    const float* vr = V + (size_t)c * N_T;
    float*       hr = h + (size_t)r * N_T;
    #pragma unroll
    for (int j = 0; j < 8; ++j) {
        const int t = lane + 32 * j;
        atomicAdd(&hr[t], a * vr[t]);
    }
}

// =====================================================================
// Final: BatchNorm(h + gat_bias) + residual 1x1 conv, all F at once
// =====================================================================
__global__ __launch_bounds__(256) void final_kernel(
    const float* __restrict__ x, const float* __restrict__ h,      // h: [F][4096][256]
    const float* __restrict__ gatb, const float* __restrict__ gamma,
    const float* __restrict__ beta, const float* __restrict__ mean,
    const float* __restrict__ var, const float* __restrict__ resw,
    const float* __restrict__ resb, float* __restrict__ out)
{
    const int idx = blockIdx.x * 256 + threadIdx.x;   // over B*T*NN
    const int n  = idx & (N_NN - 1);
    const int bt = idx >> 9;
    const int t  = bt & (N_T - 1);
    const int b  = bt >> 8;
    const size_t node = (size_t)b * N_NN + n;

    float xv[N_F];
    const float* xp = x + (size_t)idx * N_F;
    #pragma unroll
    for (int f = 0; f < N_F; ++f) xv[f] = xp[f];
    float* op = out + (size_t)idx * N_F;
    #pragma unroll
    for (int i = 0; i < N_F; ++i) {
        const float hv  = h[(size_t)i * N_NODE * N_T + node * N_T + t] + gatb[i * N_T + t];
        const float val = (hv - mean[i]) * rsqrtf(var[i] + 1e-3f) * gamma[i] + beta[i];
        float r = resb[i];
        #pragma unroll
        for (int f = 0; f < N_F; ++f) r += xv[f] * resw[f * N_F + i];
        op[i] = val + r;
    }
}

// =====================================================================
extern "C" void kernel_launch(void* const* d_in, const int* in_sizes, int n_in,
                              void* d_out, int out_size, void* d_ws, size_t ws_size,
                              hipStream_t stream)
{
    (void)in_sizes; (void)n_in; (void)out_size; (void)ws_size;
    const float* x     = (const float*)d_in[0];
    const float* caw   = (const float*)d_in[1];
    const float* cab   = (const float*)d_in[2];
    const float* cbw   = (const float*)d_in[3];
    const float* cbb   = (const float*)d_in[4];
    const float* Wq    = (const float*)d_in[5];
    const float* bq    = (const float*)d_in[6];
    const float* Wk    = (const float*)d_in[7];
    const float* bk    = (const float*)d_in[8];
    const float* Wv    = (const float*)d_in[9];
    const float* gatb  = (const float*)d_in[10];
    const float* gamma = (const float*)d_in[11];
    const float* beta  = (const float*)d_in[12];
    const float* mean  = (const float*)d_in[13];
    const float* var   = (const float*)d_in[14];
    const float* resw  = (const float*)d_in[15];
    const float* resb  = (const float*)d_in[16];
    const int*   edges = (const int*)d_in[17];
    float* out = (float*)d_out;

    char* ws = (char*)d_ws;
    size_t off = 0;
    auto walloc = [&](size_t bytes) -> void* {
        void* p = ws + off;
        off = (off + bytes + 255) & ~(size_t)255;
        return p;
    };
    unsigned short* xplanes = (unsigned short*)walloc((size_t)N_F * PLANE * 2);
    unsigned short* wtconv  = (unsigned short*)walloc((size_t)60 * WMAT * 2);
    unsigned short* wtqkv   = (unsigned short*)walloc((size_t)18 * QMAT * 2);
    unsigned short* xn      = (unsigned short*)walloc((size_t)N_NODE * N_T * 2);
    float* Qb               = (float*)walloc((size_t)N_NODE * N_T * 4);
    float* Kb               = (float*)walloc((size_t)N_NODE * N_T * 4);
    float* Vb               = (float*)walloc((size_t)N_NODE * N_T * 4);
    float* attb             = (float*)walloc((size_t)N_EE * 4);
    unsigned int* mkey      = (unsigned int*)walloc((size_t)N_NODE * 4);
    float* denom            = (float*)walloc((size_t)N_NODE * 4);
    float* hbuf             = (float*)walloc((size_t)N_F * N_NODE * N_T * 4);

    xplane_kernel<<<PLANE / 256, 256, 0, stream>>>(x, xplanes);
    wprep_kernel<<<dim3(16, 16, 60), 256, 0, stream>>>(caw, cbw, wtconv);
    qprep_kernel<<<dim3(8, 8, 18), 256, 0, stream>>>(Wq, Wk, Wv, wtqkv);

    for (int i = 0; i < N_F; ++i) {
        conv_gate_kernel<<<dim3(32, 4), 256, 0, stream>>>(
            xplanes + (size_t)i * PLANE,
            wtconv + (size_t)(i * N_KS) * WMAT,            // conv A: zz = (0*6+i)*5 + k
            wtconv + (size_t)((6 + i) * N_KS) * WMAT,      // conv B: zz = (1*6+i)*5 + k
            cab + (size_t)i * N_NN, cbb + (size_t)i * N_NN, xn);

        qkv_kernel<<<dim3(64, 2, 3), 256, 0, stream>>>(
            xn,
            wtqkv + (size_t)(0 * 6 + i) * QMAT,
            wtqkv + (size_t)(1 * 6 + i) * QMAT,
            wtqkv + (size_t)(2 * 6 + i) * QMAT,
            bq + (size_t)i * N_T, bk + (size_t)i * N_T,
            Qb, Kb, Vb);

        float* hi = hbuf + (size_t)i * N_NODE * N_T;
        gat_init_kernel<<<(N_NODE * N_T) / 256, 256, 0, stream>>>(mkey, denom, hi);
        att_kernel<<<N_EE / 8, 256, 0, stream>>>(Qb, Kb, edges, attb, mkey);
        expsum_kernel<<<N_EE / 256, 256, 0, stream>>>(attb, mkey, denom, edges);
        scatter_kernel<<<N_EE / 8, 256, 0, stream>>>(attb, denom, Vb, edges, hi);
    }

    final_kernel<<<(N_B * N_T * N_NN) / 256, 256, 0, stream>>>(
        x, hbuf, gatb, gamma, beta, mean, var, resw, resb, out);
}
